// lstm_89283780149389
// MI455X (gfx1250) — compile-verified, matrix operand order checked by reading, not run on hardware
//
#include <hip/hip_runtime.h>
#include <stdint.h>

#define T_STEPS 1024
#define BATCH   64
#define HID     512
#define H4      2048
#define K2      1024   // 2H
#define NBLK    64     // 32 hidden groups x 2 row halves
#define ROWS    32     // batch rows per block
#define ASTRIDE 136    // ushorts per LDS row (128 + 8 pad -> bank-conflict free)
#define ACHUNK  (ROWS * ASTRIDE)   // ushorts per A buffer

typedef __bf16 v16bf __attribute__((ext_vector_type(16)));
typedef float  v8f   __attribute__((ext_vector_type(8)));
typedef float  vf2   __attribute__((ext_vector_type(2)));
typedef __bf16 vbf2  __attribute__((ext_vector_type(2)));

static __device__ __forceinline__ unsigned int cvt2_bf16(float a, float b) {
  vf2 f; f[0] = a; f[1] = b;
  union { vbf2 v; unsigned int u; } x;
  x.v = __builtin_convertvector(f, vbf2);   // native packed f32->bf16 cvt
  return x.u;
}
static __device__ __forceinline__ unsigned short f2bf1(float a) {
  union { __bf16 h; unsigned short u; } x;
  x.h = (__bf16)a;
  return x.u;
}
static __device__ __forceinline__ float sigm(float z) {
  return 1.0f / (1.0f + __expf(-z));
}

// LDS byte offset of a shared-memory pointer
static __device__ __forceinline__ unsigned lds_off(const void* p) {
  return (unsigned)(unsigned long long)(__attribute__((address_space(3))) const void*)p;
}
// CDNA5 async copy: 16B global -> LDS, tracked by ASYNCcnt (no VGPR round-trip)
static __device__ __forceinline__ void async_cp16(unsigned lds_addr,
                                                  const unsigned short* g) {
  asm volatile("global_load_async_to_lds_b128 %0, %1, off"
               :: "v"(lds_addr), "v"(g) : "memory");
}
static __device__ __forceinline__ void wait_async0() {
  asm volatile("s_wait_asynccnt 0x0" ::: "memory");
}

// fp32 -> bf16 bulk convert (used for W and optionally all of X)
__global__ void cvt_f32_bf16(const float* __restrict__ src,
                             unsigned short* __restrict__ dst, int n4) {
  int idx = blockIdx.x * blockDim.x + threadIdx.x;
  if (idx < n4) {
    float4 v = ((const float4*)src)[idx];
    uint2 o;
    o.x = cvt2_bf16(v.x, v.y);
    o.y = cvt2_bf16(v.z, v.w);
    ((uint2*)dst)[idx] = o;
  }
}

__global__ void prep_state(const float* __restrict__ hidden,
                           const float* __restrict__ cell,
                           unsigned short* __restrict__ h_bf,
                           float* __restrict__ c_ws,
                           unsigned int* __restrict__ counter) {
  int idx = blockIdx.x * blockDim.x + threadIdx.x;
  if (idx == 0) *counter = 0u;                // fresh barrier state every call
  if (idx < BATCH * HID) {
    h_bf[idx] = f2bf1(hidden[idx]);
    c_ws[idx] = cell[idx];
  }
}

// Persistent recurrent kernel: 64 blocks x 256 threads (8 waves).
// Block = (hidden group hg: 16 cols) x (row half: 32 batch rows).
// Wave w: gate q = w>>1, M-tile m = w&1  -> one 16x16 z tile, K = 1024.
template <bool PRE>
__global__ __launch_bounds__(256) void lstm_persistent(
    const float* __restrict__ xf,             // [T,B,H] fp32 (used if !PRE)
    const unsigned short* __restrict__ xb,    // [T,B,H] bf16 (used if PRE)
    const float* __restrict__ bias,           // [4H]
    const unsigned short* __restrict__ WcT,   // [4H,2H] bf16 row-major
    unsigned short* __restrict__ h_bf,        // [B,H] bf16 recurrent state
    float* __restrict__ c_ws,                 // [B,H] fp32 cell state
    unsigned int* __restrict__ counter,
    float* __restrict__ out)                  // [T*B*H] ++ [B*H] c_last
{
  __shared__ unsigned short sA[2 * ACHUNK];   // double-buffered 32x128 bf16 chunk
  __shared__ float          sZ[ROWS * 66];    // z tile 32x64, padded stride

  const int tid   = threadIdx.x;
  const int lane  = tid & 31;
  const int wv    = tid >> 5;
  const int q     = wv >> 1;                  // gate 0..3 (f,i,g,o)
  const int m     = wv & 1;                   // M tile within block
  const int hg    = blockIdx.x >> 1;          // hidden group 0..31
  const int rbase = (blockIdx.x & 1) * ROWS;  // batch-row base (0 or 32)
  const int j0    = hg * 16;

  // staging: each thread copies 32B (two b128): row tid>>3, 16 k-values
  const int srow  = tid >> 3;
  const unsigned st_u = (unsigned)(srow * ASTRIDE + (tid & 7) * 16); // ushort off

  const int nrow  = q * HID + j0 + (lane & 15);    // WcT row for B fragment
  const int khalf = (lane >> 4) * 8;               // lane-half K offset
  const unsigned short* bbase = WcT + (size_t)nrow * K2 + khalf;

  const unsigned afr = (unsigned)((m * 16 + (lane & 15)) * ASTRIDE + khalf);
  const unsigned sA_base = lds_off(sA);

  // stage chunk ck of step t into buffer `buf`
  auto stage = [&](int t, int ck, int buf) {
    const unsigned laddr = sA_base + (unsigned)buf * (ACHUNK * 2) + st_u * 2;
    if (PRE || ck >= 4) {
      const unsigned short* src =
          (ck < 4) ? (xb + (size_t)t * (BATCH * HID)
                         + (rbase + srow) * HID + ck * 128 + (tid & 7) * 16)
                   : (h_bf + (rbase + srow) * HID + (ck - 4) * 128 + (tid & 7) * 16);
      async_cp16(laddr, src);
      async_cp16(laddr + 16, src + 8);
    } else {
      const float4* fs = (const float4*)(xf + (size_t)t * (BATCH * HID)
                                         + (rbase + srow) * HID + ck * 128
                                         + (tid & 7) * 16);
      __align__(16) unsigned int tmp[8];
      #pragma unroll
      for (int i = 0; i < 4; ++i) {
        float4 v = fs[i];
        tmp[i * 2]     = cvt2_bf16(v.x, v.y);
        tmp[i * 2 + 1] = cvt2_bf16(v.z, v.w);
      }
      uint4* dst = (uint4*)(sA + buf * ACHUNK + st_u);
      dst[0] = ((const uint4*)tmp)[0];
      dst[1] = ((const uint4*)tmp)[1];
    }
  };

  for (int t = 0; t < T_STEPS; ++t) {
    v8f acc = {};

    stage(t, 0, 0);
    wait_async0();
    __syncthreads();

    for (int kc = 0; kc < 8; ++kc) {
      const int cur = kc & 1;
      if (kc < 7) stage(t, kc + 1, cur ^ 1);   // prefetch next chunk into other buf

      #pragma unroll
      for (int kk = 0; kk < 4; ++kk) {
        union { uint4 qd[2]; v16bf v; } bf, af;
        const unsigned short* bp = bbase + kc * 128 + kk * 32;
        bf.qd[0] = *(const uint4*)bp;
        bf.qd[1] = *(const uint4*)(bp + 16);
        const unsigned short* ap = sA + cur * ACHUNK + afr + kk * 32;
        af.qd[0] = *(const uint4*)ap;
        af.qd[1] = *(const uint4*)(ap + 16);
        acc = __builtin_amdgcn_wmma_f32_16x16x32_bf16(false, af.v, false, bf.v,
                                                      (short)0, acc, false, false);
      }
      if (kc < 7) wait_async0();               // next chunk resident in LDS
      __syncthreads();                         // all waves done reading cur
    }

    // spill z tile to LDS (D layout: reg r, lane l -> M=r+8*(l/16), N=l%16)
    {
      const int col  = q * 16 + (lane & 15);
      const int rloc = m * 16 + (lane >> 4) * 8;
      #pragma unroll
      for (int r = 0; r < 8; ++r) sZ[(rloc + r) * 66 + col] = acc[r];
    }
    __syncthreads();

    // gates + state update: 32 rows x 16 local hidden cols
    for (int e = tid; e < ROWS * 16; e += 256) {
      const int br = e >> 4;
      const int b  = rbase + br;
      const int jl = e & 15;
      const int j  = j0 + jl;
      float zf = sZ[br * 66 + jl]      + bias[j];
      float zi = sZ[br * 66 + 16 + jl] + bias[HID + j];
      float zg = sZ[br * 66 + 32 + jl] + bias[2 * HID + j];
      float zo = sZ[br * 66 + 48 + jl] + bias[3 * HID + j];
      float fg = sigm(zf), ig = sigm(zi), gg = tanhf(zg), og = sigm(zo);
      float c = c_ws[b * HID + j];
      c = c * fg + gg * ig;
      float h = c * og;
      c_ws[b * HID + j] = c;
      h_bf[b * HID + j] = f2bf1(h);
      out[(size_t)t * (BATCH * HID) + b * HID + j] = h;
      if (t == T_STEPS - 1)
        out[(size_t)T_STEPS * BATCH * HID + b * HID + j] = c;
    }

    // device-wide step barrier (release/acquire, AGENT scope)
    __threadfence();
    __syncthreads();
    if (tid == 0) {
      __hip_atomic_fetch_add(counter, 1u, __ATOMIC_RELEASE, __HIP_MEMORY_SCOPE_AGENT);
      const unsigned int target = (unsigned)NBLK * (unsigned)(t + 1);
      while (__hip_atomic_load(counter, __ATOMIC_ACQUIRE,
                               __HIP_MEMORY_SCOPE_AGENT) < target) {
        __builtin_amdgcn_s_sleep(2);
      }
    }
    __syncthreads();
  }
}

extern "C" void kernel_launch(void* const* d_in, const int* in_sizes, int n_in,
                              void* d_out, int out_size, void* d_ws, size_t ws_size,
                              hipStream_t stream) {
  const float* input  = (const float*)d_in[0];   // [T,B,H]
  const float* hidden = (const float*)d_in[1];   // [B,H]
  const float* cell   = (const float*)d_in[2];   // [B,H]
  const float* W      = (const float*)d_in[3];   // [4H,2H]
  const float* bias   = (const float*)d_in[4];   // [4H]
  float* out = (float*)d_out;

  char* ws = (char*)d_ws;
  unsigned int*   counter = (unsigned int*)ws;                            // 256 B
  float*          c_ws    = (float*)(ws + 256);                           // 128 KB
  unsigned short* h_bf    = (unsigned short*)(ws + 256 + 131072);         // 64 KB
  unsigned short* WcT     = (unsigned short*)(ws + 256 + 131072 + 65536); // 4 MB
  unsigned short* Xb      = (unsigned short*)(ws + 256 + 131072 + 65536
                                              + (size_t)H4 * K2 * 2);     // 64 MB opt
  const size_t need_pre = 256 + 131072 + 65536 + (size_t)H4 * K2 * 2
                          + (size_t)T_STEPS * BATCH * HID * 2;
  const bool pre = ws_size >= need_pre;

  // bf16 weight image (L2-resident thereafter)
  cvt_f32_bf16<<<(H4 * K2 / 4 + 255) / 256, 256, 0, stream>>>(W, WcT, H4 * K2 / 4);
  if (pre)  // whole X to bf16 once: ~192 MB of HBM traffic, ~8 us at 23.3 TB/s
    cvt_f32_bf16<<<(T_STEPS * BATCH * HID / 4 + 255) / 256, 256, 0, stream>>>(
        input, Xb, T_STEPS * BATCH * HID / 4);
  prep_state<<<(BATCH * HID + 255) / 256, 256, 0, stream>>>(hidden, cell, h_bf,
                                                            c_ws, counter);
  if (pre)
    lstm_persistent<true><<<NBLK, 256, 0, stream>>>(input, Xb, bias, WcT, h_bf,
                                                    c_ws, counter, out);
  else
    lstm_persistent<false><<<NBLK, 256, 0, stream>>>(input, nullptr, bias, WcT,
                                                     h_bf, c_ws, counter, out);
}
